// UnifiedDualPhasorLM_37082747633932
// MI455X (gfx1250) — matrix operand order repaired and checked by reading, not compiled, not run on hardware
//
#include <hip/hip_runtime.h>
#include <hip/hip_bf16.h>
#include <math.h>

// ---------------- problem constants ----------------
#define V_  32000
#define D_  512
#define NL_ 4
#define K_  32
#define B_  2
#define L_  1024
#define FF_ 2048
#define M_  (B_ * L_)   // 2048 tokens

typedef __attribute__((ext_vector_type(16))) _Float16 v16h;
typedef __attribute__((ext_vector_type(8)))  _Float16 v8h;
typedef __attribute__((ext_vector_type(8)))  float    v8f;

// ---------------- embedding gather ----------------
__global__ void k_embed(const int* __restrict__ tokens,
                        const float* __restrict__ embed,
                        float* __restrict__ h, int total) {
    int i = blockIdx.x * blockDim.x + threadIdx.x;
    if (i >= total) return;
    int m = i / D_;
    int d = i - m * D_;
    h[i] = embed[(size_t)tokens[m] * D_ + d];
}

// ---------------- f32 -> f16 convert (embed for tied head) ----------------
__global__ void k_cvt(const float* __restrict__ in, _Float16* __restrict__ out, int n) {
    int i = blockIdx.x * blockDim.x + threadIdx.x;
    if (i < n) out[i] = (_Float16)in[i];
}

// ---------------- transpose + convert: W[Kd,N] f32 -> Wh[Npad,Kd] f16 ----------------
// grid = (Npad/32, Kd/32), block = 256 (32x8). Rows n >= N are zero-filled.
__global__ __launch_bounds__(256) void k_transcvt(const float* __restrict__ W,
                                                  _Float16* __restrict__ Wh,
                                                  int Kd, int N) {
    __shared__ float t[32][33];
    const int k0 = blockIdx.y * 32;
    const int n0 = blockIdx.x * 32;
    const int tx = threadIdx.x & 31;
    const int ty = threadIdx.x >> 5;     // 0..7
    #pragma unroll
    for (int r = 0; r < 4; ++r) {
        int k = k0 + ty + r * 8;
        int n = n0 + tx;
        t[ty + r * 8][tx] = (n < N) ? W[(size_t)k * N + n] : 0.f;
    }
    __syncthreads();
    #pragma unroll
    for (int r = 0; r < 4; ++r) {
        int n = n0 + ty + r * 8;         // output row (padded space)
        int k = k0 + tx;                 // output col
        Wh[(size_t)n * Kd + k] = (_Float16)t[tx][ty + r * 8];
    }
}

// ---------------- layernorm (one row per block of 256) ----------------
__global__ void k_layernorm(const float* __restrict__ X,
                            const float* __restrict__ g,
                            const float* __restrict__ b,
                            float* __restrict__ Yf,
                            _Float16* __restrict__ Yh, int Dd) {
    int row = blockIdx.x;
    int tid = threadIdx.x;
    const float* xr = X + (size_t)row * Dd;
    float s0 = 0.f, s1 = 0.f;
    for (int j = tid; j < Dd; j += 256) { float t = xr[j]; s0 += t; s1 += t * t; }
    __shared__ float r0[256], r1[256];
    r0[tid] = s0; r1[tid] = s1;
    __syncthreads();
    for (int st = 128; st > 0; st >>= 1) {
        if (tid < st) { r0[tid] += r0[tid + st]; r1[tid] += r1[tid + st]; }
        __syncthreads();
    }
    float mean = r0[0] / (float)Dd;
    float var  = r1[0] / (float)Dd - mean * mean;
    float inv  = rsqrtf(var + 1e-5f);
    for (int j = tid; j < Dd; j += 256) {
        float y = (xr[j] - mean) * inv * g[j] + b[j];
        if (Yf) Yf[(size_t)row * Dd + j] = y;
        if (Yh) Yh[(size_t)row * Dd + j] = (_Float16)y;
    }
}

// ---------------- WMMA GEMM: C[M,N] = A_f16[M,Kd] @ Bh_f16[N,Kd]^T + bias ----------------
// Preconditions: M % 64 == 0, Kd % 32 == 0, Bh has ceil(N/64)*64 rows (zero-padded).
// Block = 256 threads (8 wave32). BM=64, BN=64, BK=32.
// Wave w: N-tile = (w&3)*16, M-tiles = (w>>2)*32, +16  -> 2 wmma / K-step.
// Inner loop: 2x global b128 + 2x ds_store b128 + 6x ds_load b128 + 2x v_wmma.
__global__ __launch_bounds__(256) void k_gemm(const _Float16* __restrict__ A,
                                              const _Float16* __restrict__ Bh,
                                              const float* __restrict__ bias,
                                              float* __restrict__ C,
                                              int M, int N, int Kd) {
    constexpr int BM = 64, BN = 64, BK = 32;
    __shared__ _Float16 As[BM * BK];   // [m][k]
    __shared__ _Float16 Bs[BN * BK];   // [n][k]

    const int tid  = threadIdx.x;
    const int wave = tid >> 5;
    const int lane = tid & 31;
    const int tn   = (wave & 3) * 16;        // 0,16,32,48
    const int tm0  = (wave >> 2) * 32;       // 0 or 32
    const int row0 = blockIdx.y * BM;
    const int col0 = blockIdx.x * BN;

    // staging coordinates: each thread moves one 16-byte chunk of A and of B
    const int sm = tid >> 2;                 // 0..63 tile row (A) / tile col (B)
    const int sk = (tid & 3) * 8;            // 0,8,16,24

    const _Float16* Arow = A  + (size_t)(row0 + sm) * Kd + sk;
    const _Float16* Brow = Bh + (size_t)(col0 + sm) * Kd + sk;

    v8f acc0 = {}, acc1 = {};

    for (int k0 = 0; k0 < Kd; k0 += BK) {
        __syncthreads();
        *(v8h*)(&As[sm * BK + sk]) = *(const v8h*)(Arow + k0);
        *(v8h*)(&Bs[sm * BK + sk]) = *(const v8h*)(Brow + k0);
        if (k0 + BK < Kd) {
            __builtin_prefetch(Arow + k0 + BK, 0, 1);
            __builtin_prefetch(Brow + k0 + BK, 0, 1);
        }
        __syncthreads();

        // ---- fragments per ISA 7.12.2 layouts ----
        const int l15 = lane & 15;
        const int hi  = lane >> 4;

        v16h bf;
        {
            v8h lo = *(const v8h*)(&Bs[(tn + l15) * BK + 8 * hi]);
            v8h hb = *(const v8h*)(&Bs[(tn + l15) * BK + 16 + 8 * hi]);
            #pragma unroll
            for (int e = 0; e < 8; ++e) { bf[e] = lo[e]; bf[e + 8] = hb[e]; }
        }
        v16h a0;
        {
            v8h lo = *(const v8h*)(&As[(tm0 + l15) * BK + 8 * hi]);
            v8h hb = *(const v8h*)(&As[(tm0 + l15) * BK + 16 + 8 * hi]);
            #pragma unroll
            for (int e = 0; e < 8; ++e) { a0[e] = lo[e]; a0[e + 8] = hb[e]; }
        }
        v16h a1;
        {
            v8h lo = *(const v8h*)(&As[(tm0 + 16 + l15) * BK + 8 * hi]);
            v8h hb = *(const v8h*)(&As[(tm0 + 16 + l15) * BK + 16 + 8 * hi]);
            #pragma unroll
            for (int e = 0; e < 8; ++e) { a1[e] = lo[e]; a1[e + 8] = hb[e]; }
        }
        acc0 = __builtin_amdgcn_wmma_f32_16x16x32_f16(false, a0, false, bf,
                                                      (short)0, acc0, false, false);
        acc1 = __builtin_amdgcn_wmma_f32_16x16x32_f16(false, a1, false, bf,
                                                      (short)0, acc1, false, false);
    }

    // ---- epilogue: lane 0-15: M=j,N=lane; lane 16-31: M=8+j,N=lane-16 ----
    const int l15 = lane & 15;
    const int hi  = lane >> 4;
    const int col = col0 + tn + l15;
    if (col < N) {
        float bvadd = bias[col];
        #pragma unroll
        for (int j = 0; j < 8; ++j) {
            int row = row0 + tm0 + j + 8 * hi;
            C[(size_t)row * N + col] = acc0[j] + bvadd;
        }
        #pragma unroll
        for (int j = 0; j < 8; ++j) {
            int row = row0 + tm0 + 16 + j + 8 * hi;
            C[(size_t)row * N + col] = acc1[j] + bvadd;
        }
    }
}

// ---------------- elementwise ----------------
__global__ void k_tanh_h(const float* __restrict__ in, _Float16* __restrict__ out, int n) {
    int i = blockIdx.x * blockDim.x + threadIdx.x;
    if (i < n) out[i] = (_Float16)tanhf(in[i]);
}
__global__ void k_cp_pi(float* __restrict__ x, int n) {
    int i = blockIdx.x * blockDim.x + threadIdx.x;
    if (i < n) x[i] = tanhf(x[i]) * 3.14159265358979323846f;
}
__global__ void k_gelu_h(const float* __restrict__ in, _Float16* __restrict__ out, int n) {
    int i = blockIdx.x * blockDim.x + threadIdx.x;
    if (i < n) {
        float t = in[i];
        out[i] = (_Float16)(0.5f * t * (1.f + erff(t * 0.70710678118654752f)));
    }
}
__global__ void k_add3(float* __restrict__ h, const float* __restrict__ x,
                       const float* __restrict__ a, int n) {
    int i = blockIdx.x * blockDim.x + threadIdx.x;
    if (i < n) h[i] = h[i] + x[i] + a[i];
}
__global__ void k_add2(float* __restrict__ h, const float* __restrict__ a, int n) {
    int i = blockIdx.x * blockDim.x + threadIdx.x;
    if (i < n) h[i] = h[i] + a[i];
}

// ---------------- phasor memory: one wave32 per (b,d); lane = k ----------------
__global__ void k_phasor(const float* __restrict__ cp,   // [M,K]
                         const float* __restrict__ v,    // [M,D]
                         const float* __restrict__ ps_p,
                         const float* __restrict__ cs_p,
                         float* __restrict__ ret) {      // [M,D]
    int bd = blockIdx.x;         // B*D blocks
    int b  = bd / D_;
    int d  = bd - b * D_;
    int k  = threadIdx.x;        // 0..31 == K
    float ps = *ps_p, cs = *cs_p;
    float freq = __powf(10000.0f, -(float)k / (float)K_);
    float r = 0.f, im = 0.f;
    for (int l = 0; l < L_; ++l) {
        size_t m = (size_t)b * L_ + l;
        float phase = ps * ((float)l * freq) + cs * cp[m * K_ + k];
        float c = __cosf(phase);
        float s = __sinf(phase);
        float vd = v[m * D_ + d];
        r  += c * vd;
        im += s * vd;
        float t = c * r + s * im;
        #pragma unroll
        for (int off = 16; off > 0; off >>= 1) t += __shfl_down(t, off, 32);
        if (k == 0) ret[m * D_ + d] = t * rsqrtf((float)((l + 1) * K_));
    }
}

// ---------------- host orchestration ----------------
static inline dim3 gemm_grid(int M, int N) { return dim3((N + 63) / 64, (M + 63) / 64); }

extern "C" void kernel_launch(void* const* d_in, const int* in_sizes, int n_in,
                              void* d_out, int out_size, void* d_ws, size_t ws_size,
                              hipStream_t stream) {
    (void)in_sizes; (void)n_in; (void)out_size; (void)ws_size;
    const int*   tokens   = (const int*)d_in[0];
    const float* embed    = (const float*)d_in[1];
    const float* ln1_g    = (const float*)d_in[2];
    const float* ln1_b    = (const float*)d_in[3];
    const float* cp_w1    = (const float*)d_in[4];
    const float* cp_b1    = (const float*)d_in[5];
    const float* cp_w2    = (const float*)d_in[6];
    const float* cp_b2    = (const float*)d_in[7];
    const float* pos_s    = (const float*)d_in[8];
    const float* con_s    = (const float*)d_in[9];
    const float* val_w    = (const float*)d_in[10];
    const float* val_b    = (const float*)d_in[11];
    const float* oln_g    = (const float*)d_in[12];
    const float* oln_b    = (const float*)d_in[13];
    const float* out_w    = (const float*)d_in[14];
    const float* out_b    = (const float*)d_in[15];
    const float* ln2_g    = (const float*)d_in[16];
    const float* ln2_b    = (const float*)d_in[17];
    const float* ffn_w1   = (const float*)d_in[18];
    const float* ffn_b1   = (const float*)d_in[19];
    const float* ffn_w2   = (const float*)d_in[20];
    const float* ffn_b2   = (const float*)d_in[21];
    const float* no_g     = (const float*)d_in[22];
    const float* no_b     = (const float*)d_in[23];
    const float* head_b   = (const float*)d_in[24];
    float* logits = (float*)d_out;

    // ---- scratch carving ----
    char* ws = (char*)d_ws;
    size_t off = 0;
    auto carve = [&](size_t bytes) -> void* {
        void* p = ws + off;
        off += (bytes + 255) & ~(size_t)255;
        return p;
    };
    // activations
    float*    h     = (float*)   carve((size_t)M_ * D_ * 4);
    float*    xf32  = (float*)   carve((size_t)M_ * D_ * 4);
    _Float16* xf16  = (_Float16*)carve((size_t)M_ * D_ * 2);
    float*    gbuf  = (float*)   carve((size_t)M_ * FF_ * 4);
    float*    cpbuf = (float*)   carve((size_t)M_ * K_ * 4);
    float*    vbuf  = (float*)   carve((size_t)M_ * D_ * 4);
    float*    ret   = (float*)   carve((size_t)M_ * D_ * 4);
    _Float16* h16a  = (_Float16*)carve((size_t)M_ * FF_ * 2);
    _Float16* h16b  = (_Float16*)carve((size_t)M_ * FF_ * 2);
    // f16 transposed weights ([Npad, Kd] each)
    const int KP = 64;  // cp_w2 N=32 padded to 64
    _Float16* w1h  = (_Float16*)carve((size_t)NL_ * D_  * D_  * 2);  // [D,D]
    _Float16* w2h  = (_Float16*)carve((size_t)NL_ * KP  * D_  * 2);  // [64,D]
    _Float16* vwh  = (_Float16*)carve((size_t)NL_ * D_  * D_  * 2);
    _Float16* owh  = (_Float16*)carve((size_t)NL_ * D_  * D_  * 2);
    _Float16* fw1h = (_Float16*)carve((size_t)NL_ * FF_ * D_  * 2);  // [FF,D]
    _Float16* fw2h = (_Float16*)carve((size_t)NL_ * D_  * FF_ * 2);  // [D,FF]
    _Float16* embh = (_Float16*)carve((size_t)V_  * D_  * 2);        // [V,D] (already [N,Kd])

    const int EW = 256;
    const int nMD = M_ * D_, nMK = M_ * K_, nMF = M_ * FF_;

    // ---- one-time weight conversion (f32 -> f16, transposed to [N,Kd]) ----
    k_cvt<<<(V_ * D_ + EW - 1) / EW, EW, 0, stream>>>(embed, embh, V_ * D_);
    for (int i = 0; i < NL_; ++i) {
        k_transcvt<<<dim3(D_ / 32, D_ / 32), 256, 0, stream>>>(
            cp_w1 + (size_t)i * D_ * D_,  w1h  + (size_t)i * D_ * D_,  D_, D_);
        k_transcvt<<<dim3(KP / 32, D_ / 32), 256, 0, stream>>>(
            cp_w2 + (size_t)i * D_ * K_,  w2h  + (size_t)i * KP * D_,  D_, K_);
        k_transcvt<<<dim3(D_ / 32, D_ / 32), 256, 0, stream>>>(
            val_w + (size_t)i * D_ * D_,  vwh  + (size_t)i * D_ * D_,  D_, D_);
        k_transcvt<<<dim3(D_ / 32, D_ / 32), 256, 0, stream>>>(
            out_w + (size_t)i * D_ * D_,  owh  + (size_t)i * D_ * D_,  D_, D_);
        k_transcvt<<<dim3(FF_ / 32, D_ / 32), 256, 0, stream>>>(
            ffn_w1 + (size_t)i * D_ * FF_, fw1h + (size_t)i * FF_ * D_, D_, FF_);
        k_transcvt<<<dim3(D_ / 32, FF_ / 32), 256, 0, stream>>>(
            ffn_w2 + (size_t)i * FF_ * D_, fw2h + (size_t)i * D_ * FF_, FF_, D_);
    }

    // ---- embedding ----
    k_embed<<<(nMD + EW - 1) / EW, EW, 0, stream>>>(tokens, embed, h, nMD);

    for (int i = 0; i < NL_; ++i) {
        const _Float16* w1  = w1h  + (size_t)i * D_ * D_;
        const _Float16* w2  = w2h  + (size_t)i * KP * D_;
        const _Float16* vw  = vwh  + (size_t)i * D_ * D_;
        const _Float16* ow  = owh  + (size_t)i * D_ * D_;
        const _Float16* fw1 = fw1h + (size_t)i * FF_ * D_;
        const _Float16* fw2 = fw2h + (size_t)i * D_ * FF_;
        const float* b1  = cp_b1  + (size_t)i * D_;
        const float* b2  = cp_b2  + (size_t)i * K_;
        const float* vb  = val_b  + (size_t)i * D_;
        const float* ob  = out_b  + (size_t)i * D_;
        const float* fb1 = ffn_b1 + (size_t)i * FF_;
        const float* fb2 = ffn_b2 + (size_t)i * D_;

        // x = LN1(h)
        k_layernorm<<<M_, 256, 0, stream>>>(h, ln1_g + (size_t)i * D_, ln1_b + (size_t)i * D_,
                                            xf32, xf16, D_);
        // t1 = tanh(x @ cp_w1 + b1)
        k_gemm<<<gemm_grid(M_, D_), 256, 0, stream>>>(xf16, w1, b1, gbuf, M_, D_, D_);
        k_tanh_h<<<(nMD + EW - 1) / EW, EW, 0, stream>>>(gbuf, h16a, nMD);
        // cp = tanh(t1 @ cp_w2 + b2) * pi
        k_gemm<<<gemm_grid(M_, K_), 256, 0, stream>>>(h16a, w2, b2, cpbuf, M_, K_, D_);
        k_cp_pi<<<(nMK + EW - 1) / EW, EW, 0, stream>>>(cpbuf, nMK);
        // v = x @ val_w + vb
        k_gemm<<<gemm_grid(M_, D_), 256, 0, stream>>>(xf16, vw, vb, vbuf, M_, D_, D_);
        // phasor memory -> retrieved
        k_phasor<<<B_ * D_, 32, 0, stream>>>(cpbuf, vbuf, pos_s + i, con_s + i, ret);
        // attn = LN_o(retrieved) @ out_w + ob ; h = h + x + attn
        k_layernorm<<<M_, 256, 0, stream>>>(ret, oln_g + (size_t)i * D_, oln_b + (size_t)i * D_,
                                            nullptr, h16a, D_);
        k_gemm<<<gemm_grid(M_, D_), 256, 0, stream>>>(h16a, ow, ob, vbuf, M_, D_, D_);
        k_add3<<<(nMD + EW - 1) / EW, EW, 0, stream>>>(h, xf32, vbuf, nMD);
        // FFN
        k_layernorm<<<M_, 256, 0, stream>>>(h, ln2_g + (size_t)i * D_, ln2_b + (size_t)i * D_,
                                            nullptr, h16a, D_);
        k_gemm<<<gemm_grid(M_, FF_), 256, 0, stream>>>(h16a, fw1, fb1, gbuf, M_, FF_, D_);
        k_gelu_h<<<(nMF + EW - 1) / EW, EW, 0, stream>>>(gbuf, h16b, nMF);
        k_gemm<<<gemm_grid(M_, D_), 256, 0, stream>>>(h16b, fw2, fb2, vbuf, M_, D_, FF_);
        k_add2<<<(nMD + EW - 1) / EW, EW, 0, stream>>>(h, vbuf, nMD);
    }

    // logits = LN(h) @ embed^T + head_b   (embh is [V,D] f16 == [N,Kd])
    k_layernorm<<<M_, 256, 0, stream>>>(h, no_g, no_b, nullptr, h16a, D_);
    k_gemm<<<gemm_grid(M_, V_), 256, 0, stream>>>(h16a, embh, head_b, logits, M_, V_, D_);
}
